// AttnNet_26379689132174
// MI455X (gfx1250) — matrix operand order repaired and checked by reading, not compiled, not run on hardware
//
#include <hip/hip_runtime.h>
#include <hip/hip_bf16.h>

typedef __bf16 v16bf __attribute__((ext_vector_type(16)));
typedef __bf16 v8bf  __attribute__((ext_vector_type(8)));
typedef float  v8f   __attribute__((ext_vector_type(8)));
typedef unsigned int u32x4 __attribute__((ext_vector_type(4)));
typedef int          i32x8 __attribute__((ext_vector_type(8)));
typedef int          i32x4 __attribute__((ext_vector_type(4)));

#define WMMA_BF16(a, b, c) \
  __builtin_amdgcn_wmma_f32_16x16x32_bf16(false, (a), false, (b), (short)0, (c), false, false)

// ---------------- sizes ----------------
#define Bz 64
#define Tz 256
#define Ez 300
#define EP 320
#define Hz 512
#define Dz 1024
#define G4 2048
#define OUTz 256
#define ROWS (Bz*Tz)

// ---------------- scratch layout (bytes) ----------------
#define OFF_EMB       ((size_t)0)                            // [16384][320] bf16
#define OFF_XPRE_F    (OFF_EMB + (size_t)ROWS*EP*2)          // time-major [t][g][hcol][b] bf16
#define OFF_XPRE_B    (OFF_XPRE_F + (size_t)ROWS*G4*2)
#define OFF_CTX       (OFF_XPRE_B + (size_t)ROWS*G4*2)       // [16384][1024] bf16
#define OFF_CTXP      (OFF_CTX + (size_t)ROWS*Dz*2)          // ctx packed as B-fragments per batch
#define OFF_TARGET    (OFF_CTXP + (size_t)ROWS*Dz*2)
#define OFF_SCORES    (OFF_TARGET + (size_t)ROWS*Dz*2)       // f32 [64][256][256]
#define OFF_ATTN      (OFF_SCORES + (size_t)Bz*Tz*Tz*4)      // bf16
#define OFF_WEIGHTED  (OFF_ATTN + (size_t)Bz*Tz*Tz*2)        // [16384][1024] bf16
#define OFF_WIHP_F    (OFF_WEIGHTED + (size_t)ROWS*Dz*2)     // frag-packed [10][128][32][16]
#define OFF_WIHP_B    (OFF_WIHP_F + (size_t)EP*G4*2)
#define OFF_WHHP_F    (OFF_WIHP_B + (size_t)EP*G4*2)         // [16 hb][16 kk][4 g][2 nt][32][16]
#define OFF_WHHP_B    (OFF_WHHP_F + (size_t)Hz*G4*2)
#define OFF_WINP      (OFF_WHHP_B + (size_t)Hz*G4*2)         // [32][64][32][16]
#define OFF_WOAP      (OFF_WINP + (size_t)Dz*Dz*2)           // [64][64][32][16]
#define OFF_WOUTP     (OFF_WOAP + (size_t)(2*Dz)*Dz*2)       // [32][16][32][16]
#define OFF_HBUF      (OFF_WOUTP + (size_t)Dz*OUTz*2)        // 2 dir * 2 slot * [64][512] bf16
#define OFF_CTXOUT    (OFF_HBUF + (size_t)2*2*Bz*Hz*2)       // f32 [64][1024]
#define OFF_CTXOUT_BF (OFF_CTXOUT + (size_t)Bz*Dz*4)
#define OFF_Y         (OFF_CTXOUT_BF + (size_t)Bz*Dz*2)      // f32 [64][256]
#define OFF_BARRIER   (OFF_Y + (size_t)Bz*OUTz*4)

// ---------------- fragment loaders ----------------
// A 16x32 bf16 from row-major: two contiguous 16B loads per lane
__device__ __forceinline__ v16bf load_a_frag(const __bf16* A, int lda, int mt, int kk, int lane) {
  const int m  = mt * 16 + (lane & 15);
  const int kb = kk * 32 + ((lane >> 4) << 4);
  const __bf16* p = A + (size_t)m * lda + kb;
  v16bf a;
#pragma unroll
  for (int e = 0; e < 8; ++e) { a[e] = p[e]; a[e + 8] = p[e + 8]; }
  return a;
}
// B fragment from pre-packed fragment-major storage: one aligned 32B load per lane
__device__ __forceinline__ v16bf load_b_packed(const __bf16* Bp, int Ntiles, int nt, int kk, int lane) {
  return *(const v16bf*)(Bp + ((size_t)((kk * Ntiles + nt) * 32 + lane) << 4));
}
// B fragment from [N][K] storage (contiguous along K)
__device__ __forceinline__ v16bf load_bt_frag(const __bf16* Bt, int ldk, int nt, int kk, int lane) {
  const int n  = nt * 16 + (lane & 15);
  const int kb = kk * 32 + ((lane >> 4) << 4);
  const __bf16* p = Bt + (size_t)n * ldk + kb;
  v16bf b;
#pragma unroll
  for (int e = 0; e < 16; ++e) b[e] = p[e];
  return b;
}

// ---------------- weight pack: f32 src [N][K] -> frag-major bf16 ----------------
__global__ void k_pack_w(const float* __restrict__ src, __bf16* __restrict__ dst,
                         int N, int K, int Ntiles, long total) {
  long i = (long)blockIdx.x * 256 + threadIdx.x;
  if (i >= total) return;
  int e = (int)(i & 15), lane = (int)((i >> 4) & 31);
  long rest = i >> 9;
  int nt = (int)(rest % Ntiles), kk = (int)(rest / Ntiles);
  int n = nt * 16 + (lane & 15);
  int k = kk * 32 + ((lane >> 4) << 4) + e;
  dst[i] = (k < K) ? (__bf16)src[(size_t)n * K + k] : (__bf16)0.0f;
}

// w_hh [2048][512] -> [16 hb][16 kk][4 g][2 nt][32 lane][16 e]
__global__ void k_pack_whh(const float* __restrict__ src, __bf16* __restrict__ dst) {
  int i = blockIdx.x * 256 + threadIdx.x;           // 1048576 total
  int e = i & 15, lane = (i >> 4) & 31, nt = (i >> 9) & 1;
  int g = (i >> 10) & 3, kk = (i >> 12) & 15, hb = i >> 16;
  int n = g * Hz + hb * 32 + nt * 16 + (lane & 15);
  int k = kk * 32 + ((lane >> 4) << 4) + e;
  dst[i] = (__bf16)src[(size_t)n * Hz + k];
}

// ctx [b*256+k][1024] bf16 -> per-batch frag-major [b][8 kk][64 nt][32][16]
__global__ void k_pack_ctx(const __bf16* __restrict__ ctx, __bf16* __restrict__ dst) {
  int i = blockIdx.x * 256 + threadIdx.x;           // 16777216 total
  int e = i & 15, lane = (i >> 4) & 31, nt = (i >> 9) & 63;
  int kk = (i >> 15) & 7, b = i >> 18;
  int k = kk * 32 + ((lane >> 4) << 4) + e;
  int n = nt * 16 + (lane & 15);
  dst[i] = ctx[((size_t)b * Tz + k) * Dz + n];
}

// ---------------- embedding gather ----------------
__global__ void k_embed(const int* __restrict__ idx, const float* __restrict__ table,
                        __bf16* __restrict__ out) {
  int r = blockIdx.x, e = threadIdx.x;
  int tok = idx[r];
  out[(size_t)r * EP + e] = (e < Ez) ? (__bf16)table[(size_t)tok * Ez + e] : (__bf16)0.0f;
}

// ---------------- x_pre GEMM: emb[16384,320] @ wihp -> time-major layout ----------------
__global__ void k_gemm_xpre(const __bf16* __restrict__ A, const __bf16* __restrict__ Bp,
                            const float* __restrict__ bias0, const float* __restrict__ bias1,
                            __bf16* __restrict__ xpre_t) {
  int wt = blockIdx.x * 8 + (threadIdx.x >> 5);     // Mtiles=1024 x NQ=32
  int mt = wt >> 5, ntq = wt & 31, lane = threadIdx.x & 31;
  v8f acc[4];
#pragma unroll
  for (int j = 0; j < 4; ++j) acc[j] = (v8f){0.f,0.f,0.f,0.f,0.f,0.f,0.f,0.f};
  for (int kk = 0; kk < EP / 32; ++kk) {
    v16bf a = load_a_frag(A, EP, mt, kk, lane);
#pragma unroll
    for (int j = 0; j < 4; ++j)
      acc[j] = WMMA_BF16(a, load_b_packed(Bp, 128, ntq * 4 + j, kk, lane), acc[j]);
  }
#pragma unroll
  for (int j = 0; j < 4; ++j) {
    int n = (ntq * 4 + j) * 16 + (lane & 15);
    int g = n >> 9, hcol = n & 511;
    float bv = bias0[n] + bias1[n];
#pragma unroll
    for (int r = 0; r < 8; ++r) {
      int m = mt * 16 + r + 8 * (lane >> 4);
      int b = m >> 8, t = m & 255;
      xpre_t[((size_t)(t * 4 + g) * Hz + hcol) * Bz + b] = (__bf16)(acc[j][r] + bv);
    }
  }
}

// ---------------- persistent bidirectional LSTM; whh slice staged in LDS via TDM ----------------
__global__ void k_lstm(const __bf16* __restrict__ xpre_t_f, const __bf16* __restrict__ xpre_t_b,
                       const __bf16* __restrict__ whhp_f, const __bf16* __restrict__ whhp_b,
                       __bf16* __restrict__ hbuf, __bf16* __restrict__ ctx,
                       unsigned* __restrict__ barrier) {
  extern __shared__ __bf16 lds_whh[];               // 65536 halves = 128 KB
  const int dir = blockIdx.x >> 4;
  const int hb  = blockIdx.x & 15;
  const int lane = threadIdx.x & 31;
  const int wave = threadIdx.x >> 5;
  const int mt = wave >> 1;                         // 16-row batch tile
  const int nt = wave & 1;                          // 16-col slice of this WG's 32 hidden units
  const __bf16* xpre = dir ? xpre_t_b : xpre_t_f;
  const __bf16* whhp = (dir ? whhp_b : whhp_f) + (size_t)hb * 65536;
  __bf16* hping = hbuf + (size_t)dir * (2 * Bz * Hz);
  unsigned* cnt = barrier + dir;
  const int hcol = hb * 32 + nt * 16 + (lane & 15);
  const int mrow8 = mt * 16 + 8 * (lane >> 4);

  if (wave == 0) {
    // Tensor Data Mover: copy this WG's 128KB recurrent-weight slice into LDS.
    // D#: 2-byte elems, 2D tile 4096 x 16, contiguous (stride 4096).
    unsigned lds_off = (unsigned)(unsigned long long)(uintptr_t)&lds_whh[0];
    unsigned long long ga = (unsigned long long)(uintptr_t)whhp;
    u32x4 g0;
    g0[0] = 1u;                                           // count=1
    g0[1] = lds_off;                                      // lds_addr
    g0[2] = (unsigned)ga;                                 // global_addr[31:0]
    g0[3] = (unsigned)((ga >> 32) & 0x1FFFFFFu) | (2u << 30);  // addr[56:32] | type=2
    i32x8 g1;
    g1[0] = 0x00010000;                                   // data_size = 2 bytes
    g1[1] = (int)(4096u << 16);                           // tensor_dim0 = 4096
    g1[2] = (int)(16u << 16);                             // tensor_dim1 = 16
    g1[3] = (int)(4096u << 16);                           // tile_dim0 = 4096
    g1[4] = 16;                                           // tile_dim1 = 16
    g1[5] = 4096;                                         // tensor_dim0_stride = 4096
    g1[6] = 0; g1[7] = 0;
    i32x4 gz4 = {0, 0, 0, 0};
    i32x8 gz8 = {0, 0, 0, 0, 0, 0, 0, 0};
    __builtin_amdgcn_tensor_load_to_lds(g0, g1, gz4, gz4, gz8, 0);
    __builtin_amdgcn_s_wait_tensorcnt(0);
  }
  __syncthreads();

  v8f c = {0.f,0.f,0.f,0.f,0.f,0.f,0.f,0.f};
  for (int t = 0; t < Tz; ++t) {
    const int tix = dir ? (Tz - 1 - t) : t;
    const __bf16* hprev = hping + (size_t)(t & 1) * (Bz * Hz);
    v8f acc[4];
#pragma unroll
    for (int g = 0; g < 4; ++g) {
      v8bf xv = *(const v8bf*)(xpre + ((size_t)(tix * 4 + g) * Hz + hcol) * Bz + mrow8);
#pragma unroll
      for (int r = 0; r < 8; ++r) acc[g][r] = (float)xv[r];
    }
    for (int kk = 0; kk < Hz / 32; ++kk) {
      v16bf a = load_a_frag(hprev, Hz, mt, kk, lane);
#pragma unroll
      for (int g = 0; g < 4; ++g) {
        v16bf bf = *(const v16bf*)(lds_whh + ((size_t)(((kk * 4 + g) * 2 + nt) * 32 + lane) << 4));
        acc[g] = WMMA_BF16(a, bf, acc[g]);
      }
    }
#pragma unroll
    for (int r = 0; r < 8; ++r) {
      float ig = 1.f / (1.f + __expf(-acc[0][r]));
      float fg = 1.f / (1.f + __expf(-acc[1][r]));
      float gg = tanhf(acc[2][r]);
      float og = 1.f / (1.f + __expf(-acc[3][r]));
      c[r] = fg * c[r] + ig * gg;
      float h = og * tanhf(c[r]);
      int m = mrow8 + r;
      __bf16 h16 = (__bf16)h;
      hping[(size_t)((t + 1) & 1) * (Bz * Hz) + (size_t)m * Hz + hcol] = h16;
      ctx[((size_t)m * Tz + tix) * Dz + dir * Hz + hcol] = h16;
    }
    __syncthreads();
    if (threadIdx.x == 0) {
      __threadfence();
      atomicAdd(cnt, 1u);
      unsigned target = 16u * (unsigned)(t + 1);
      while (atomicAdd(cnt, 0u) < target) { __builtin_amdgcn_s_sleep(2); }
    }
    __syncthreads();
    __threadfence();
  }
}

// ---------------- generic GEMM, packed B, 4 N-tiles per wave, bf16 out ----------------
__global__ void k_gemm_bf16(const __bf16* __restrict__ A, int lda, long sA,
                            const __bf16* __restrict__ Bp, int Ntiles, long sB,
                            __bf16* __restrict__ C, int ldc, long sC,
                            int Mtiles, int Ksteps) {
  A += (size_t)blockIdx.y * sA; Bp += (size_t)blockIdx.y * sB; C += (size_t)blockIdx.y * sC;
  int NQ = Ntiles >> 2;
  int wt = blockIdx.x * 8 + (threadIdx.x >> 5);
  if (wt >= Mtiles * NQ) return;
  int mt = wt / NQ, ntq = wt % NQ, lane = threadIdx.x & 31;
  v8f acc[4];
#pragma unroll
  for (int j = 0; j < 4; ++j) acc[j] = (v8f){0.f,0.f,0.f,0.f,0.f,0.f,0.f,0.f};
  for (int kk = 0; kk < Ksteps; ++kk) {
    v16bf a = load_a_frag(A, lda, mt, kk, lane);
#pragma unroll
    for (int j = 0; j < 4; ++j)
      acc[j] = WMMA_BF16(a, load_b_packed(Bp, Ntiles, ntq * 4 + j, kk, lane), acc[j]);
  }
#pragma unroll
  for (int j = 0; j < 4; ++j) {
    int n = (ntq * 4 + j) * 16 + (lane & 15);
#pragma unroll
    for (int r = 0; r < 8; ++r) {
      int m = mt * 16 + r + 8 * (lane >> 4);
      C[(size_t)m * ldc + n] = (__bf16)acc[j][r];
    }
  }
}

// ---------------- scores: per-batch target @ ctx^T ----------------
__global__ void k_gemm_scores(const __bf16* __restrict__ target, const __bf16* __restrict__ ctx,
                              float* __restrict__ scores) {
  const int b = blockIdx.y;
  const __bf16* A  = target + (size_t)b * Tz * Dz;
  const __bf16* Bt = ctx    + (size_t)b * Tz * Dz;
  float* S = scores + (size_t)b * Tz * Tz;
  int wt = blockIdx.x * 8 + (threadIdx.x >> 5);     // Mtiles=16 x NQ=4
  int mt = wt >> 2, ntq = wt & 3, lane = threadIdx.x & 31;
  v8f acc[4];
#pragma unroll
  for (int j = 0; j < 4; ++j) acc[j] = (v8f){0.f,0.f,0.f,0.f,0.f,0.f,0.f,0.f};
  for (int kk = 0; kk < Dz / 32; ++kk) {
    v16bf a = load_a_frag(A, Dz, mt, kk, lane);
#pragma unroll
    for (int j = 0; j < 4; ++j)
      acc[j] = WMMA_BF16(a, load_bt_frag(Bt, Dz, ntq * 4 + j, kk, lane), acc[j]);
  }
#pragma unroll
  for (int j = 0; j < 4; ++j) {
    int n = (ntq * 4 + j) * 16 + (lane & 15);
#pragma unroll
    for (int r = 0; r < 8; ++r) {
      int m = mt * 16 + r + 8 * (lane >> 4);
      S[(size_t)m * Tz + n] = acc[j][r];
    }
  }
}

// ---------------- masked softmax -> bf16 attn ----------------
__global__ void k_softmax(const float* __restrict__ scores, const unsigned char* __restrict__ mask,
                          __bf16* __restrict__ attn) {
  const int row = blockIdx.x;
  const int b = row >> 8;
  const int k = threadIdx.x;
  __shared__ float red[Tz];
  float v = mask[b * Tz + k] ? -__builtin_inff() : scores[(size_t)row * Tz + k];
  red[k] = v; __syncthreads();
  for (int s = 128; s > 0; s >>= 1) { if (k < s) red[k] = fmaxf(red[k], red[k + s]); __syncthreads(); }
  float mx = red[0]; __syncthreads();
  float e = __expf(v - mx);
  red[k] = e; __syncthreads();
  for (int s = 128; s > 0; s >>= 1) { if (k < s) red[k] += red[k + s]; __syncthreads(); }
  attn[(size_t)row * Tz + k] = (__bf16)(e / red[0]);
}

// ---------------- h_tilde = tanh(concat(weighted,ctx) @ woa^T); fused mean over T ----------------
__global__ void k_gemm_htilde(const __bf16* __restrict__ weighted, const __bf16* __restrict__ ctx,
                              const __bf16* __restrict__ woap, float* __restrict__ ctx_out) {
  int wt = blockIdx.x * 8 + (threadIdx.x >> 5);     // Mtiles=1024 x NQ=16
  int mt = wt >> 4, ntq = wt & 15, lane = threadIdx.x & 31;
  v8f acc[4];
#pragma unroll
  for (int j = 0; j < 4; ++j) acc[j] = (v8f){0.f,0.f,0.f,0.f,0.f,0.f,0.f,0.f};
  for (int kk = 0; kk < 64; ++kk) {                 // K = 2048
    const __bf16* Asrc = (kk < 32) ? weighted : ctx;
    v16bf a = load_a_frag(Asrc, Dz, mt, kk & 31, lane);
#pragma unroll
    for (int j = 0; j < 4; ++j)
      acc[j] = WMMA_BF16(a, load_b_packed(woap, 64, ntq * 4 + j, kk, lane), acc[j]);
  }
#pragma unroll
  for (int j = 0; j < 4; ++j) {
    int n = (ntq * 4 + j) * 16 + (lane & 15);
#pragma unroll
    for (int r = 0; r < 8; ++r) {
      int m = mt * 16 + r + 8 * (lane >> 4);
      atomicAdd(&ctx_out[(size_t)(m >> 8) * Dz + n], tanhf(acc[j][r]) * (1.0f / Tz));
    }
  }
}

__global__ void k_cvt_bf16(const float* __restrict__ src, __bf16* __restrict__ dst, int n) {
  int i = blockIdx.x * 256 + threadIdx.x;
  if (i < n) dst[i] = (__bf16)src[i];
}

// ---------------- y = tanh(ctx_out @ wout^T + b_out) ----------------
__global__ void k_gemm_y(const __bf16* __restrict__ A, const __bf16* __restrict__ woutp,
                         const float* __restrict__ b_out, float* __restrict__ y) {
  int wt = blockIdx.x * 8 + (threadIdx.x >> 5);     // Mtiles=4 x NQ=4
  int mt = wt >> 2, ntq = wt & 3, lane = threadIdx.x & 31;
  v8f acc[4];
#pragma unroll
  for (int j = 0; j < 4; ++j) acc[j] = (v8f){0.f,0.f,0.f,0.f,0.f,0.f,0.f,0.f};
  for (int kk = 0; kk < Dz / 32; ++kk) {
    v16bf a = load_a_frag(A, Dz, mt, kk, lane);
#pragma unroll
    for (int j = 0; j < 4; ++j)
      acc[j] = WMMA_BF16(a, load_b_packed(woutp, 16, ntq * 4 + j, kk, lane), acc[j]);
  }
#pragma unroll
  for (int j = 0; j < 4; ++j) {
    int n = (ntq * 4 + j) * 16 + (lane & 15);
#pragma unroll
    for (int r = 0; r < 8; ++r) {
      int m = mt * 16 + r + 8 * (lane >> 4);
      y[(size_t)m * OUTz + n] = tanhf(acc[j][r] + b_out[n]);
    }
  }
}

// ---------------- BatchNorm1d (batch stats, biased var) ----------------
__global__ void k_batchnorm(const float* __restrict__ y, const float* __restrict__ gamma,
                            const float* __restrict__ beta, float* __restrict__ out) {
  int f = threadIdx.x;
  float mu = 0.f;
  for (int r = 0; r < Bz; ++r) mu += y[r * OUTz + f];
  mu *= (1.0f / Bz);
  float var = 0.f;
  for (int r = 0; r < Bz; ++r) { float d = y[r * OUTz + f] - mu; var += d * d; }
  var *= (1.0f / Bz);
  float inv = rsqrtf(var + 1e-5f);
  for (int r = 0; r < Bz; ++r)
    out[r * OUTz + f] = gamma[f] * (y[r * OUTz + f] - mu) * inv + beta[f];
}

extern "C" void kernel_launch(void* const* d_in, const int* in_sizes, int n_in,
                              void* d_out, int out_size, void* d_ws, size_t ws_size,
                              hipStream_t stream) {
  const int*    inputs = (const int*)d_in[0];
  const unsigned char* mask = (const unsigned char*)d_in[1];
  const float*  table  = (const float*)d_in[2];
  const float*  w_ih_f = (const float*)d_in[3];
  const float*  w_hh_f = (const float*)d_in[4];
  const float*  b_ih_f = (const float*)d_in[5];
  const float*  b_hh_f = (const float*)d_in[6];
  const float*  w_ih_b = (const float*)d_in[7];
  const float*  w_hh_b = (const float*)d_in[8];
  const float*  b_ih_b = (const float*)d_in[9];
  const float*  b_hh_b = (const float*)d_in[10];
  const float*  w_in   = (const float*)d_in[11];
  const float*  w_oa   = (const float*)d_in[12];
  const float*  w_out  = (const float*)d_in[13];
  const float*  b_out  = (const float*)d_in[14];
  const float*  gamma  = (const float*)d_in[15];
  const float*  beta   = (const float*)d_in[16];

  char* ws = (char*)d_ws;
  __bf16* emb      = (__bf16*)(ws + OFF_EMB);
  __bf16* xpre_f   = (__bf16*)(ws + OFF_XPRE_F);
  __bf16* xpre_b   = (__bf16*)(ws + OFF_XPRE_B);
  __bf16* ctx      = (__bf16*)(ws + OFF_CTX);
  __bf16* ctxp     = (__bf16*)(ws + OFF_CTXP);
  __bf16* target   = (__bf16*)(ws + OFF_TARGET);
  float*  scores   = (float*)(ws + OFF_SCORES);
  __bf16* attn     = (__bf16*)(ws + OFF_ATTN);
  __bf16* weighted = (__bf16*)(ws + OFF_WEIGHTED);
  __bf16* wihp_f   = (__bf16*)(ws + OFF_WIHP_F);
  __bf16* wihp_b   = (__bf16*)(ws + OFF_WIHP_B);
  __bf16* whhp_f   = (__bf16*)(ws + OFF_WHHP_F);
  __bf16* whhp_b   = (__bf16*)(ws + OFF_WHHP_B);
  __bf16* winp     = (__bf16*)(ws + OFF_WINP);
  __bf16* woap     = (__bf16*)(ws + OFF_WOAP);
  __bf16* woutp    = (__bf16*)(ws + OFF_WOUTP);
  __bf16* hbuf     = (__bf16*)(ws + OFF_HBUF);
  float*  ctx_out  = (float*)(ws + OFF_CTXOUT);
  __bf16* ctx_out_bf = (__bf16*)(ws + OFF_CTXOUT_BF);
  float*  y        = (float*)(ws + OFF_Y);
  unsigned* barrier = (unsigned*)(ws + OFF_BARRIER);

  (void)hipMemsetAsync(barrier, 0, 256, stream);
  (void)hipMemsetAsync(hbuf, 0, (size_t)2 * 2 * Bz * Hz * 2, stream);
  (void)hipMemsetAsync(ctx_out, 0, (size_t)Bz * Dz * 4, stream);

  auto tg = [](long n) { return (unsigned)((n + 255) / 256); };
  // fragment-major weight packs
  k_pack_w<<<tg((long)EP * G4), 256, 0, stream>>>(w_ih_f, wihp_f, G4, Ez, 128, (long)EP * G4);
  k_pack_w<<<tg((long)EP * G4), 256, 0, stream>>>(w_ih_b, wihp_b, G4, Ez, 128, (long)EP * G4);
  k_pack_whh<<<tg((long)Hz * G4), 256, 0, stream>>>(w_hh_f, whhp_f);
  k_pack_whh<<<tg((long)Hz * G4), 256, 0, stream>>>(w_hh_b, whhp_b);
  k_pack_w<<<tg((long)Dz * Dz), 256, 0, stream>>>(w_in, winp, Dz, Dz, 64, (long)Dz * Dz);
  k_pack_w<<<tg((long)2 * Dz * Dz), 256, 0, stream>>>(w_oa, woap, Dz, 2 * Dz, 64, (long)2 * Dz * Dz);
  k_pack_w<<<tg((long)Dz * OUTz), 256, 0, stream>>>(w_out, woutp, OUTz, Dz, 16, (long)Dz * OUTz);

  k_embed<<<ROWS, EP, 0, stream>>>(inputs, table, emb);

  // x_pre (time-major) for both directions: M=16384, N=2048, K=320
  k_gemm_xpre<<<(1024 * 32) / 8, 256, 0, stream>>>(emb, wihp_f, b_ih_f, b_hh_f, xpre_f);
  k_gemm_xpre<<<(1024 * 32) / 8, 256, 0, stream>>>(emb, wihp_b, b_ih_b, b_hh_b, xpre_b);

  // persistent recurrence: 128KB dynamic LDS per WG for TDM-staged weights
  k_lstm<<<32, 256, 131072, stream>>>(xpre_f, xpre_b, whhp_f, whhp_b, hbuf, ctx, barrier);

  // target = ctx @ w_in^T : M=16384, N=1024, K=1024
  k_gemm_bf16<<<(1024 * 16) / 8, 256, 0, stream>>>(ctx, Dz, 0, winp, 64, 0,
      target, Dz, 0, 1024, Dz / 32);

  k_gemm_scores<<<dim3(8, Bz), 256, 0, stream>>>(target, ctx, scores);
  k_softmax<<<ROWS, Tz, 0, stream>>>(scores, mask, attn);

  // weighted = attn @ ctx (batched): repack ctx as B-fragments, then GEMM
  k_pack_ctx<<<tg((long)ROWS * Dz), 256, 0, stream>>>(ctx, ctxp);
  k_gemm_bf16<<<dim3((16 * 16) / 8, Bz), 256, 0, stream>>>(attn, Tz, (long)Tz * Tz,
      ctxp, 64, (long)Tz * Dz, weighted, Dz, (long)Tz * Dz, 16, Tz / 32);

  k_gemm_htilde<<<(1024 * 16) / 8, 256, 0, stream>>>(weighted, ctx, woap, ctx_out);

  k_cvt_bf16<<<tg(Bz * Dz), 256, 0, stream>>>(ctx_out, ctx_out_bf, Bz * Dz);
  k_gemm_y<<<(4 * 4) / 8, 256, 0, stream>>>(ctx_out_bf, woutp, b_out, y);
  k_batchnorm<<<1, OUTz, 0, stream>>>(y, gamma, beta, (float*)d_out);
}
